// CNN_GAT_52604759441725
// MI455X (gfx1250) — compile-verified, matrix operand order checked by reading, not run on hardware
//
#include <hip/hip_runtime.h>
#include <hip/hip_bf16.h>

// ---------------- constants (match reference) ----------------
#define BB   16
#define LL   384
#define NN   128
#define FF   256
#define HH   4
#define CC   64
#define NEG_SLOPE 0.2f
#define EPSN 1e-5f

typedef __attribute__((ext_vector_type(16))) _Float16 v16h;
typedef __attribute__((ext_vector_type(8)))  float    v8f;
typedef __attribute__((ext_vector_type(4))) unsigned int u32x4;
typedef __attribute__((ext_vector_type(8))) int          i32x8;
typedef __attribute__((ext_vector_type(4))) int          i32x4;

#define DEVINL __device__ __forceinline__

// ---- WMMA fragment helpers (CDNA5 16x16x32 f16, wave32 layouts) ----
// A (16x32 f16): lanes 0-15 row M=lane, elems 0..7 -> K 0..7, 8..15 -> K 16..23;
// lanes 16-31: same rows, K 8..15 / 24..31.  B mirrors with col = lane&15.
DEVINL v16h load_frag(const _Float16* base, int stride, int lane) {
  const int hi = (lane >> 4) & 1;
  const int r  = lane & 15;
  const int kb = hi * 8;
  const _Float16* p = base + r * stride;
  v16h f;
#pragma unroll
  for (int i = 0; i < 8; ++i) f[i] = p[kb + i];
#pragma unroll
  for (int i = 0; i < 8; ++i) f[8 + i] = p[16 + kb + i];
  return f;
}

// same layout, converting f32 source on the fly (for the alpha matrix)
DEVINL v16h load_frag_f32(const float* base, int stride, int lane) {
  const int hi = (lane >> 4) & 1;
  const int r  = lane & 15;
  const int kb = hi * 8;
  const float* p = base + r * stride;
  v16h f;
#pragma unroll
  for (int i = 0; i < 8; ++i) f[i] = (_Float16)p[kb + i];
#pragma unroll
  for (int i = 0; i < 8; ++i) f[8 + i] = (_Float16)p[16 + kb + i];
  return f;
}

DEVINL v8f wmma_f16(v16h a, v16h b, v8f c) {
  return __builtin_amdgcn_wmma_f32_16x16x32_f16(false, a, false, b,
                                                (short)0, c, false, false);
}

// monotone order-preserving float<->uint encoding for atomicMax segment-max
DEVINL unsigned fenc(float x) {
  unsigned u = __float_as_uint(x);
  return (u & 0x80000000u) ? ~u : (u | 0x80000000u);
}
DEVINL float fdec(unsigned u) {
  return __uint_as_float((u & 0x80000000u) ? (u ^ 0x80000000u) : ~u);
}

// ---------------- weight prep ----------------
__global__ void k_wprep(const float* __restrict__ src, _Float16* __restrict__ dst,
                        int Co, int Ci) {
  int idx = blockIdx.x * blockDim.x + threadIdx.x;
  int tot = Co * Ci * 3;
  if (idx >= tot) return;
  int co  = idx / (Ci * 3);
  int rem = idx - co * Ci * 3;
  int ci  = rem / 3;
  int d   = rem - ci * 3;
  dst[d * Co * Ci + co * Ci + ci] = (_Float16)src[idx];
}

__global__ void k_cvt(const float* __restrict__ src, _Float16* __restrict__ dst, int n) {
  int idx = blockIdx.x * blockDim.x + threadIdx.x;
  if (idx < n) dst[idx] = (_Float16)src[idx];
}

// ---------------- fused CNN conv layer: 3 shifted GEMMs ----------------
template <int CIN, int COUT>
DEVINL void conv_gemm(const _Float16* act, _Float16* outa,
                      const _Float16* __restrict__ wgt,   // [3][COUT][CIN]
                      const float* __restrict__ bias, int tid) {
  const int wave = tid >> 5, lane = tid & 31;
  const int hi = lane >> 4, r = lane & 15;
  const int NT = LL / 16, TOT = (COUT / 16) * NT;
  for (int tt = wave; tt < TOT; tt += 8) {
    const int mt = tt / NT, nt = tt - mt * NT;
    if (tt + 8 < TOT) __builtin_prefetch(wgt + ((tt + 8) / NT) * 16 * CIN, 0, 1);
    v8f acc = {};
#pragma unroll
    for (int d = 0; d < 3; ++d) {
      const _Float16* wd = wgt + d * COUT * CIN + (mt * 16) * CIN;
      const _Float16* xd = act + (nt * 16 + d) * CIN;   // row (t+d-1)+pad
#pragma unroll
      for (int kc = 0; kc < CIN; kc += 32) {
        v16h a = load_frag(wd + kc, CIN, lane);
        v16h b = load_frag(xd + kc, CIN, lane);
        acc = wmma_f16(a, b, acc);
      }
    }
    const int t = nt * 16 + r;
#pragma unroll
    for (int v = 0; v < 8; ++v) {
      const int co = mt * 16 + hi * 8 + v;
      float y = acc[v] + bias[co];
      outa[(t + 1) * COUT + co] = (_Float16)(y > 0.f ? y : 0.f);
    }
  }
}

template <int CIN, int COUT>
DEVINL void conv_gemm_pool(const _Float16* act, float* pool,
                           const _Float16* __restrict__ wgt,
                           const float* __restrict__ bias, int tid) {
  const int wave = tid >> 5, lane = tid & 31;
  const int hi = lane >> 4;
  const int NT = LL / 16, TOT = (COUT / 16) * NT;
  for (int tt = wave; tt < TOT; tt += 8) {
    const int mt = tt / NT, nt = tt - mt * NT;
    if (tt + 8 < TOT) __builtin_prefetch(wgt + ((tt + 8) / NT) * 16 * CIN, 0, 1);
    v8f acc = {};
#pragma unroll
    for (int d = 0; d < 3; ++d) {
      const _Float16* wd = wgt + d * COUT * CIN + (mt * 16) * CIN;
      const _Float16* xd = act + (nt * 16 + d) * CIN;
#pragma unroll
      for (int kc = 0; kc < CIN; kc += 32) {
        v16h a = load_frag(wd + kc, CIN, lane);
        v16h b = load_frag(xd + kc, CIN, lane);
        acc = wmma_f16(a, b, acc);
      }
    }
#pragma unroll
    for (int v = 0; v < 8; ++v) {
      const int co = mt * 16 + hi * 8 + v;
      float y = acc[v] + bias[co];
      if (y > 0.f) atomicAdd(&pool[co], y);   // ds_add_f32
    }
  }
}

// ------------- kernel 1: fused CNN + mean pool + FC, one WG per sequence ----
__global__ __launch_bounds__(256)
void k_cnn(const float* __restrict__ xg,
           const float* __restrict__ w1, const float* __restrict__ b1,
           const _Float16* __restrict__ w2h, const float* __restrict__ b2,
           const _Float16* __restrict__ w3h, const float* __restrict__ b3,
           const _Float16* __restrict__ w4h, const float* __restrict__ b4,
           const float* __restrict__ fcw, const float* __restrict__ fcb,
           float* __restrict__ node) {
  extern __shared__ unsigned char smem[];
  float*    xs   = (float*)smem;                                   // 386 f32 (byte 0)
  _Float16* a1   = (_Float16*)(smem + 1544);                       // 386*32
  _Float16* a2   = a1 + 386 * 32;                                  // 386*64
  _Float16* a3   = a2 + 386 * 64;                                  // 386*128
  float*    pool = (float*)(a3 + 386 * 128);                       // 256
  float*    feat = pool + 256;                                     // 256

  const int tid = threadIdx.x;
  const int s = blockIdx.x;            // sequence id = b*N + n
  const int b = s >> 7, n = s & (NN - 1);

  // zero halo rows + pool
  for (int c = tid; c < 32;  c += 256) { a1[c] = (_Float16)0.f; a1[385 * 32  + c] = (_Float16)0.f; }
  for (int c = tid; c < 64;  c += 256) { a2[c] = (_Float16)0.f; a2[385 * 64  + c] = (_Float16)0.f; }
  for (int c = tid; c < 128; c += 256) { a3[c] = (_Float16)0.f; a3[385 * 128 + c] = (_Float16)0.f; }
  pool[tid] = 0.f;
  if (tid == 0) { xs[0] = 0.f; xs[385] = 0.f; }

  // --- stage this node's time series (strided column of x[B,L,N]) into LDS ---
#if __has_builtin(__builtin_amdgcn_tensor_load_to_lds) && __has_builtin(__builtin_amdgcn_s_wait_tensorcnt)
  if (tid < 32) {   // one wave issues the TDM descriptor (values are uniform)
    const unsigned long long ga =
        (unsigned long long)(size_t)(xg + (size_t)b * LL * NN + n);
    u32x4 g0;
    g0[0] = 1u;                                   // count=1 (valid user D#)
    g0[1] = 4u;                                   // lds_addr: &xs[1] (dyn-LDS base 0)
    g0[2] = (unsigned)(ga & 0xffffffffu);         // global_addr[31:0]
    g0[3] = (unsigned)((ga >> 32) & 0x1ffffffu)   // global_addr[56:32]
          | (2u << 30);                           // type = 2 ("image")
    i32x8 g1;
    g1[0] = (int)(2u << 16);                      // wg_mask=0, data_size=4B
    g1[1] = (int)(1u << 16);                      // tensor_dim0 = 1 (lo16)
    g1[2] = (int)(((unsigned)LL & 0xffffu) << 16);// dim0 hi | tensor_dim1 = 384 lo
    g1[3] = (int)(1u << 16);                      // dim1 hi | tile_dim0 = 1
    g1[4] = (int)LL;                              // tile_dim1 = 384, tile_dim2 = 0
    g1[5] = (int)NN;                              // tensor_dim0_stride = 128 elems
    g1[6] = 0; g1[7] = 0;                         // dim1_stride unused (2D)
    i32x4 z4 = {0, 0, 0, 0};
#if defined(__clang_major__) && (__clang_major__ >= 23)
    i32x8 z8 = {};
    __builtin_amdgcn_tensor_load_to_lds(g0, g1, z4, z4, z8, 0);
#else
    __builtin_amdgcn_tensor_load_to_lds(g0, g1, z4, z4, 0);
#endif
    __builtin_amdgcn_s_wait_tensorcnt(0);
  }
#else
  for (int t = tid; t < LL; t += 256) xs[t + 1] = xg[((size_t)b * LL + t) * NN + n];
#endif
  __syncthreads();

  // conv1: Cin=1 -> 32, plain VALU
  for (int idx = tid; idx < LL * 32; idx += 256) {
    const int t = idx >> 5, co = idx & 31;
    float y = b1[co];
#pragma unroll
    for (int d = 0; d < 3; ++d) y += w1[co * 3 + d] * xs[t + d];
    a1[(t + 1) * 32 + co] = (_Float16)(y > 0.f ? y : 0.f);
  }
  __syncthreads();

  conv_gemm<32, 64>(a1, a2, w2h, b2, tid);
  __syncthreads();
  conv_gemm<64, 128>(a2, a3, w3h, b3, tid);
  __syncthreads();
  conv_gemm_pool<128, 256>(a3, pool, w4h, b4, tid);
  __syncthreads();

  feat[tid] = pool[tid] * (1.f / (float)LL);
  __syncthreads();

  float acc = fcb[tid];
  for (int c = 0; c < FF; ++c) acc += fcw[tid * FF + c] * feat[c];
  node[(size_t)s * FF + tid] = acc;
}

// ------------- kernel 2: one GAT layer + GraphNorm + residual, WG per graph --
// Aggregation is dense: alpha scattered into A_h[dst][src], then WMMA GEMM.
__global__ __launch_bounds__(256)
void k_gat(const float* __restrict__ hin, float* __restrict__ hout,
           const _Float16* __restrict__ Wh,       // [256 out][256 in] f16
           const float* __restrict__ a_s, const float* __restrict__ a_d,
           const float* __restrict__ bias,
           const float* __restrict__ gam, const float* __restrict__ bet,
           const float* __restrict__ msc,
           const int* __restrict__ ei, int E) {
  extern __shared__ unsigned char smem[];
  _Float16* xh   = (_Float16*)smem;                 // 128*256 f16 (reused as Af32)
  float*    Af32 = (float*)smem;                    // 128*128 f32 (after GEMM1)
  _Float16* hhT  = xh + NN * FF;                    // [256 feat][128 node] f16
  float*    outb = (float*)(hhT + NN * FF);         // [128 node][256 feat] f32
  float*    alS  = outb + NN * FF;                  // 128*4
  float*    alD  = alS + NN * HH;                   // 128*4
  unsigned* mx   = (unsigned*)(alD + NN * HH);      // 128*4
  float*    ssum = (float*)(mx + NN * HH);          // 128*4

  const int tid = threadIdx.x;
  const int wave = tid >> 5, lane = tid & 31;
  const int hi = lane >> 4, r = lane & 15;
  const int b = blockIdx.x;
  const float* xin = hin + (size_t)b * NN * FF;

  for (int i = tid; i < NN * FF; i += 256) {
    xh[i]   = (_Float16)xin[i];
    outb[i] = bias[i & (FF - 1)];      // segment_sum target starts at +bias
  }
  for (int i = tid; i < NN * HH; i += 256) { mx[i] = 0u; ssum[i] = 0.f; }
  __syncthreads();

  // GEMM1: h = x @ W^T, stored transposed hhT[feature][node]
  const int NT = FF / 16, TOT = (NN / 16) * NT;
  for (int tt = wave; tt < TOT; tt += 8) {
    const int mt = tt / NT, nt = tt - mt * NT;
    if (tt + 8 < TOT) __builtin_prefetch(Wh + ((tt + 8) % NT) * 16 * FF, 0, 1);
    v8f acc = {};
#pragma unroll
    for (int kc = 0; kc < FF; kc += 32) {
      v16h a  = load_frag(xh + (mt * 16) * FF + kc, FF, lane);
      v16h bf = load_frag(Wh + (nt * 16) * FF + kc, FF, lane);
      acc = wmma_f16(a, bf, acc);
    }
    const int f = nt * 16 + r;                // D col = output feature
#pragma unroll
    for (int v = 0; v < 8; ++v) {            // D rows = node, contiguous store
      hhT[f * NN + mt * 16 + hi * 8 + v] = (_Float16)acc[v];
    }
  }
  __syncthreads();

  // per-node per-head attention logits from hhT
  for (int idx = tid; idx < NN * HH; idx += 256) {
    const int nn = idx >> 2, h = idx & 3;
    float s1 = 0.f, s2 = 0.f;
#pragma unroll 4
    for (int c = 0; c < CC; ++c) {
      const float hv = (float)hhT[(h * CC + c) * NN + nn];
      s1 += hv * a_s[h * CC + c];
      s2 += hv * a_d[h * CC + c];
    }
    alS[idx] = s1; alD[idx] = s2;
  }
  __syncthreads();

  const int ET = E + NN;   // + self loops
  // pass 1: segment max over dst
  for (int e = tid; e < ET; e += 256) {
    int sN, dN;
    if (e < E) { sN = ei[e]; dN = ei[E + e]; } else { sN = dN = e - E; }
#pragma unroll
    for (int h = 0; h < HH; ++h) {
      float v = alS[sN * HH + h] + alD[dN * HH + h];
      v = v > 0.f ? v : NEG_SLOPE * v;
      atomicMax(&mx[dN * HH + h], fenc(v));
    }
  }
  __syncthreads();
  // pass 2: segment sum of exp
  for (int e = tid; e < ET; e += 256) {
    int sN, dN;
    if (e < E) { sN = ei[e]; dN = ei[E + e]; } else { sN = dN = e - E; }
#pragma unroll
    for (int h = 0; h < HH; ++h) {
      float v = alS[sN * HH + h] + alD[dN * HH + h];
      v = v > 0.f ? v : NEG_SLOPE * v;
      atomicAdd(&ssum[dN * HH + h], __expf(v - fdec(mx[dN * HH + h])));
    }
  }
  __syncthreads();

  // per head: scatter alpha into dense A_h[dst][src], then out_h = A_h @ h_h
  for (int h = 0; h < HH; ++h) {
    for (int i = tid; i < NN * NN; i += 256) Af32[i] = 0.f;
    __syncthreads();
    for (int e = tid; e < ET; e += 256) {
      int sN, dN;
      if (e < E) { sN = ei[e]; dN = ei[E + e]; } else { sN = dN = e - E; }
      float v = alS[sN * HH + h] + alD[dN * HH + h];
      v = v > 0.f ? v : NEG_SLOPE * v;
      const float alpha = __expf(v - fdec(mx[dN * HH + h])) /
                          (ssum[dN * HH + h] + 1e-16f);
      atomicAdd(&Af32[dN * NN + sN], alpha);    // duplicate edges accumulate
    }
    __syncthreads();
    // GEMM2: [128 dst x 128 src] @ [128 src x 64 feat]
    const int TOT2 = (NN / 16) * (CC / 16);     // 32 tiles
    for (int tt = wave; tt < TOT2; tt += 8) {
      const int mt = tt >> 2, nt = tt & 3;
      v8f acc = {};
#pragma unroll
      for (int kc = 0; kc < NN; kc += 32) {
        v16h a  = load_frag_f32(Af32 + (mt * 16) * NN + kc, NN, lane);
        v16h bf = load_frag(hhT + (h * CC + nt * 16) * NN + kc, NN, lane);
        acc = wmma_f16(a, bf, acc);
      }
      const int f = h * CC + nt * 16 + r;
#pragma unroll
      for (int v = 0; v < 8; ++v) {
        const int m = mt * 16 + hi * 8 + v;     // dst node
        outb[m * FF + f] += acc[v];             // unique writer; bias pre-added
      }
    }
    __syncthreads();
  }

  // GraphNorm + residual ReLU; thread tid owns feature f = tid
  {
    const int f = tid;
    float mu = 0.f;
    for (int nn = 0; nn < NN; ++nn) mu += outb[nn * FF + f];
    mu *= (1.f / (float)NN);
    const float mm = msc[f] * mu;
    float var = 0.f;
    for (int nn = 0; nn < NN; ++nn) {
      const float o = outb[nn * FF + f] - mm;
      var += o * o;
    }
    var *= (1.f / (float)NN);
    const float ri = rsqrtf(var + EPSN);
    const float g = gam[f], be = bet[f];
    for (int nn = 0; nn < NN; ++nn) {
      const float o = outb[nn * FF + f] - mm;
      float y = g * o * ri + be + xin[nn * FF + f];
      hout[(size_t)b * NN * FF + nn * FF + f] = y > 0.f ? y : 0.f;
    }
  }
}

// ------------- kernel 3: mean pool + classifier ----------------
__global__ __launch_bounds__(256)
void k_cls(const float* __restrict__ hin, const float* __restrict__ cw,
           const float* __restrict__ cb, float* __restrict__ out) {
  __shared__ float red[256];
  const int b = blockIdx.x, f = threadIdx.x;
  float p = 0.f;
  for (int nn = 0; nn < NN; ++nn) p += hin[(size_t)b * NN * FF + nn * FF + f];
  red[f] = p * (1.f / (float)NN) * cw[f];
  __syncthreads();
  for (int s = 128; s > 0; s >>= 1) {
    if (f < s) red[f] += red[f + s];
    __syncthreads();
  }
  if (f == 0) out[b] = red[0] + cb[0];
}

// ---------------- host: orchestrate ----------------
extern "C" void kernel_launch(void* const* d_in, const int* in_sizes, int n_in,
                              void* d_out, int out_size, void* d_ws, size_t ws_size,
                              hipStream_t stream) {
  const float* x   = (const float*)d_in[0];
  const int*   ei  = (const int*)d_in[1];
  const float* c1w = (const float*)d_in[2];  const float* c1b = (const float*)d_in[3];
  const float* c2w = (const float*)d_in[4];  const float* c2b = (const float*)d_in[5];
  const float* c3w = (const float*)d_in[6];  const float* c3b = (const float*)d_in[7];
  const float* c4w = (const float*)d_in[8];  const float* c4b = (const float*)d_in[9];
  const float* fcw = (const float*)d_in[10]; const float* fcb = (const float*)d_in[11];
  const float* gw[3]  = {(const float*)d_in[12], (const float*)d_in[19], (const float*)d_in[26]};
  const float* gas[3] = {(const float*)d_in[13], (const float*)d_in[20], (const float*)d_in[27]};
  const float* gad[3] = {(const float*)d_in[14], (const float*)d_in[21], (const float*)d_in[28]};
  const float* gb[3]  = {(const float*)d_in[15], (const float*)d_in[22], (const float*)d_in[29]};
  const float* ng[3]  = {(const float*)d_in[16], (const float*)d_in[23], (const float*)d_in[30]};
  const float* nb[3]  = {(const float*)d_in[17], (const float*)d_in[24], (const float*)d_in[31]};
  const float* nms[3] = {(const float*)d_in[18], (const float*)d_in[25], (const float*)d_in[32]};
  const float* clw = (const float*)d_in[33]; const float* clb = (const float*)d_in[34];
  const int E = in_sizes[1] / 2;

  char* ws = (char*)d_ws;
  _Float16* w2h = (_Float16*)(ws + 0);          // 12288 B
  _Float16* w3h = (_Float16*)(ws + 12288);      // 49152 B
  _Float16* w4h = (_Float16*)(ws + 61440);      // 196608 B
  _Float16* g1h = (_Float16*)(ws + 258048);     // 131072 B
  _Float16* g2h = (_Float16*)(ws + 389120);
  _Float16* g3h = (_Float16*)(ws + 520192);
  float* nodeA  = (float*)(ws + 651264);        // 2 MB
  float* nodeB  = (float*)(ws + 651264 + 2097152);

  // weight prep
  k_wprep<<<(64 * 32 * 3 + 255) / 256, 256, 0, stream>>>(c2w, w2h, 64, 32);
  k_wprep<<<(128 * 64 * 3 + 255) / 256, 256, 0, stream>>>(c3w, w3h, 128, 64);
  k_wprep<<<(256 * 128 * 3 + 255) / 256, 256, 0, stream>>>(c4w, w4h, 256, 128);
  k_cvt<<<(FF * FF + 255) / 256, 256, 0, stream>>>(gw[0], g1h, FF * FF);
  k_cvt<<<(FF * FF + 255) / 256, 256, 0, stream>>>(gw[1], g2h, FF * FF);
  k_cvt<<<(FF * FF + 255) / 256, 256, 0, stream>>>(gw[2], g3h, FF * FF);

  // fused CNN + pool + FC: one WG per (b, n) sequence
  const unsigned smem1 = 1544 + (386u * (32 + 64 + 128)) * 2 + 2 * 256 * 4;  // 176,520 B
  k_cnn<<<BB * NN, 256, smem1, stream>>>(x, c1w, c1b, w2h, c2b, w3h, c3b,
                                         w4h, c4b, fcw, fcb, nodeA);

  // 3 GAT layers, one WG per graph
  const unsigned smem2 = (unsigned)(2u * NN * FF * 2 + NN * FF * 4 + 4u * NN * HH * 4); // 270,336 B
  const _Float16* gh[3] = {g1h, g2h, g3h};
  float* bufs[4] = {nodeA, nodeB, nodeA, nodeB};
  for (int l = 0; l < 3; ++l) {
    k_gat<<<BB, 256, smem2, stream>>>(bufs[l], bufs[l + 1], gh[l], gas[l], gad[l],
                                      gb[l], ng[l], nb[l], nms[l], ei, E);
  }

  k_cls<<<BB, 256, 0, stream>>>(bufs[3], clw, clb, (float*)d_out);
}
